// TverskyAttentionBackbone_62586263437956
// MI455X (gfx1250) — compile-verified, matrix operand order checked by reading, not run on hardware
//
#include <hip/hip_runtime.h>
#include <math.h>

// ---------------------------------------------------------------------------
// TverskyAttentionBackbone forward for MI455X (gfx1250, wave32, WMMA).
// Dense GEMMs: v_wmma_f32_16x16x32_f16, 2x2 register-blocked (32x32 per wave)
// so each A/B fragment feeds two WMMAs (16 FLOP/B from L2 vs 8 for 1x1).
// Tversky attention scores (L1-of-ReLU, not bilinear) stay a VALU kernel.
// Working set ~38MB << 192MB L2, so b128 global loads feed the MMA directly.
// ---------------------------------------------------------------------------

typedef __attribute__((ext_vector_type(16))) _Float16 v16h;
typedef __attribute__((ext_vector_type(8)))  _Float16 v8h;
typedef __attribute__((ext_vector_type(8)))  float    v8f;

#define NB     4
#define E      768
#define E3     2304
#define DFF    3072
#define NHD    12
#define DH     64
#define NTOK   197
#define NPAT   196
#define MROWS  (NB*NTOK)   // 788
#define MPAD   800         // 50 tiles of 16 (even)
#define MPATCH 784         // valid patch rows (alloc padded to MPAD)
#define NOUT   1000
#define NOUTP  1024        // 64 tiles of 16 (even)
#define MHEAD  32          // 2 tiles of 16 (even)

#define CEILDIV(a,b) (((a)+(b)-1)/(b))

// ----------------------------- utility kernels -----------------------------

__global__ void k_fill0(unsigned int* p, size_t nwords) {
  size_t i  = (size_t)blockIdx.x * blockDim.x + threadIdx.x;
  size_t st = (size_t)gridDim.x * blockDim.x;
  for (; i < nwords; i += st) p[i] = 0u;
}

// x:(B,C,224,224) -> A:(784, 768) f16, row = b*196+patch, col = c*256+py*16+qx
__global__ void k_im2col(const float* __restrict__ x, _Float16* __restrict__ A) {
  int idx = blockIdx.x * 256 + threadIdx.x;
  if (idx >= MPATCH * E) return;
  int row = idx / E, col = idx % E;
  int b = row / NPAT, pp = row % NPAT;
  int ph = pp / 14, pw = pp % 14;
  int c = col / 256, rem = col % 256;
  int py = rem / 16, qx = rem % 16;
  size_t xi = (((size_t)b * 3 + c) * 224 + (ph * 16 + py)) * 224 + (pw * 16 + qx);
  A[idx] = (_Float16)x[xi];
}

__global__ void k_cvt(const float* __restrict__ s, _Float16* __restrict__ d, int n) {
  int i = blockIdx.x * 256 + threadIdx.x;
  if (i < n) d[i] = (_Float16)s[i];
}

// W:[K][N] f32 -> Wt:[Npad][K] f16 (zero-pad rows n>=N)
__global__ void k_wt_t(const float* __restrict__ W, _Float16* __restrict__ Wt,
                       int K, int N, int total) {
  int i = blockIdx.x * 256 + threadIdx.x;
  if (i >= total) return;
  int n = i / K, k = i % K;
  Wt[i] = (n < N) ? (_Float16)W[(size_t)k * N + n] : (_Float16)0.f;
}

// ------------------------------- WMMA GEMM ---------------------------------
// C[Mtiles*16][Npad] = A[.][K] @ Wt[Npad][K]^T  (+bias) (+resid) (act)
// One wave computes a 32x32 macro-tile (2x2 WMMA tiles, frag reuse x2);
// 4 waves/block span 128 columns. grid = (ceil(Npad/32/4), Mtiles/2).
__device__ __forceinline__ v16h load_frag(const _Float16* row, int k0, int hs) {
  // 16-bit frag layout: elems 0..7 <- K=k0+hs*8.. ; elems 8..15 <- K=k0+16+hs*8..
  v8h lo = *(const v8h*)(row + k0 + hs * 8);
  v8h hi = *(const v8h*)(row + k0 + 16 + hs * 8);
  return __builtin_shufflevector(lo, hi, 0,1,2,3,4,5,6,7,8,9,10,11,12,13,14,15);
}

__global__ void k_gemm_wmma(const _Float16* __restrict__ A,
                            const _Float16* __restrict__ Wt,
                            const float* __restrict__ bias, int Nreal,
                            float* __restrict__ outF, _Float16* __restrict__ outH,
                            const float* __restrict__ resid,
                            int K, int Npad, int act) {
  const int Ntiles2 = Npad >> 5;       // pairs of 16-col tiles
  const int wave = threadIdx.x >> 5;
  const int lane = threadIdx.x & 31;
  const int tn2 = blockIdx.x * 4 + wave;
  const int tm2 = blockIdx.y;
  if (tn2 >= Ntiles2) return;          // wave-uniform: EXEC stays all-1 for WMMA
  const int hs  = lane >> 4;           // half-select (K-group)
  const int l16 = lane & 15;
  const _Float16* arow0 = A  + (size_t)(tm2 * 32 + l16) * K;
  const _Float16* arow1 = arow0 + (size_t)16 * K;
  const _Float16* brow0 = Wt + (size_t)(tn2 * 32 + l16) * K;
  const _Float16* brow1 = brow0 + (size_t)16 * K;
  v8f c00 = {0.f,0.f,0.f,0.f,0.f,0.f,0.f,0.f};
  v8f c01 = c00, c10 = c00, c11 = c00;
  for (int k0 = 0; k0 < K; k0 += 32) {
    v16h a0 = load_frag(arow0, k0, hs);
    v16h a1 = load_frag(arow1, k0, hs);
    v16h b0 = load_frag(brow0, k0, hs);
    v16h b1 = load_frag(brow1, k0, hs);
    c00 = __builtin_amdgcn_wmma_f32_16x16x32_f16(false, a0, false, b0, (short)0, c00, false, false);
    c01 = __builtin_amdgcn_wmma_f32_16x16x32_f16(false, a0, false, b1, (short)0, c01, false, false);
    c10 = __builtin_amdgcn_wmma_f32_16x16x32_f16(false, a1, false, b0, (short)0, c10, false, false);
    c11 = __builtin_amdgcn_wmma_f32_16x16x32_f16(false, a1, false, b1, (short)0, c11, false, false);
  }
  // epilogue: 2x2 tiles of 16x16
  auto store_tile = [&](const v8f& acc, int mbase, int ncol) {
    const float bv = (bias != nullptr && ncol < Nreal) ? bias[ncol] : 0.f;
#pragma unroll
    for (int r = 0; r < 8; ++r) {
      const int row = mbase + r + 8 * hs;  // C/D layout: VGPR r -> M = r + 8*(lane/16)
      const size_t idx = (size_t)row * Npad + ncol;
      float v = acc[r] + bv;
      if (resid) v += resid[idx];
      if (act == 1) {                      // tanh-approx GELU (jax default)
        const float c0 = 0.7978845608028654f;
        v = 0.5f * v * (1.0f + tanhf(c0 * (v + 0.044715f * v * v * v)));
      }
      if (outF) outF[idx] = v;
      if (outH) outH[idx] = (_Float16)v;
    }
  };
  const int m0 = tm2 * 32, n0 = tn2 * 32 + l16;
  store_tile(c00, m0,      n0);
  store_tile(c01, m0,      n0 + 16);
  store_tile(c10, m0 + 16, n0);
  store_tile(c11, m0 + 16, n0 + 16);
}

// ------------------------- sequence assembly & LN --------------------------

__global__ void k_assemble(const float* __restrict__ tokF, const float* __restrict__ pos,
                           const float* __restrict__ cls, float* __restrict__ h) {
  int idx = blockIdx.x * 256 + threadIdx.x;
  if (idx >= MROWS * E) return;
  int r = idx / E, c = idx % E;
  int b = r / NTOK, t = r % NTOK;
  float v;
  if (t == 0) v = cls[c];
  else {
    int p = t - 1;
    v = tokF[(size_t)(b * NPAT + p) * E + c] + pos[(size_t)p * E + c];
  }
  h[idx] = v;
}

__global__ void k_ln(const float* __restrict__ x, const float* __restrict__ w,
                     const float* __restrict__ b, _Float16* __restrict__ o) {
  const int row = blockIdx.x;
  const int tid = threadIdx.x;          // 256 threads, 3 elems each (768)
  const float* xr = x + (size_t)row * E;
  __shared__ float sr[256];
  float x0 = xr[tid], x1 = xr[tid + 256], x2 = xr[tid + 512];
  sr[tid] = x0 + x1 + x2;
  __syncthreads();
  for (int s = 128; s > 0; s >>= 1) { if (tid < s) sr[tid] += sr[tid + s]; __syncthreads(); }
  float mu = sr[0] * (1.f / E);
  __syncthreads();
  float d0 = x0 - mu, d1 = x1 - mu, d2 = x2 - mu;
  sr[tid] = d0 * d0 + d1 * d1 + d2 * d2;
  __syncthreads();
  for (int s = 128; s > 0; s >>= 1) { if (tid < s) sr[tid] += sr[tid + s]; __syncthreads(); }
  float inv = rsqrtf(sr[0] * (1.f / E) + 1e-5f);
  _Float16* orow = o + (size_t)row * E;
  orow[tid]       = (_Float16)(d0 * inv * w[tid]       + b[tid]);
  orow[tid + 256] = (_Float16)(d1 * inv * w[tid + 256] + b[tid + 256]);
  orow[tid + 512] = (_Float16)(d2 * inv * w[tid + 512] + b[tid + 512]);
}

// ---------------------------- Tversky attention ----------------------------
// grid = (197, 12, 4); block = 256. qkv:[MPAD][2304] f32 (q|k|v thirds).
__global__ void k_attn(const float* __restrict__ qkv, _Float16* __restrict__ outh) {
  const int i = blockIdx.x, h = blockIdx.y, b = blockIdx.z;
  const int tid = threadIdx.x;
  __shared__ float s_q[DH];
  __shared__ float s_attn[256];
  __shared__ float s_red[256];

  const size_t rowi = (size_t)(b * NTOK + i);
  const float* qrow = qkv + rowi * E3 + h * DH;
  if (tid < DH) s_q[tid] = fmaxf(qrow[tid], 0.f);
  __syncthreads();

  float qs = 0.f;
  for (int d = 0; d < DH; ++d) qs += s_q[d];

  float score = -3.0e38f;
  if (tid < NTOK) {
    const float* krow = qkv + (size_t)(b * NTOK + tid) * E3 + E + h * DH;
    float l1 = 0.f, ks = 0.f;
    for (int d = 0; d < DH; ++d) {
      float kp = fmaxf(krow[d], 0.f);
      ks += kp;
      l1 += fabsf(s_q[d] - kp);
    }
    float inter = 0.5f * (qs + ks - l1);
    float amb   = 0.5f * (l1 + qs - ks);
    float bma   = 0.5f * (l1 - qs + ks);
    score = 1.0f * inter / (inter + 0.5f * amb + 0.5f * bma + 1e-8f);
  }
  // softmax over j = 0..196
  s_red[tid] = score;
  __syncthreads();
  for (int s = 128; s > 0; s >>= 1) { if (tid < s) s_red[tid] = fmaxf(s_red[tid], s_red[tid + s]); __syncthreads(); }
  float mx = s_red[0];
  __syncthreads();
  float p = (tid < NTOK) ? expf(score - mx) : 0.f;
  s_red[tid] = p;
  __syncthreads();
  for (int s = 128; s > 0; s >>= 1) { if (tid < s) s_red[tid] += s_red[tid + s]; __syncthreads(); }
  float inv = 1.f / s_red[0];
  __syncthreads();
  s_attn[tid] = p * inv;
  __syncthreads();

  // o[i,d] = sum_j attn[j] * v[j,d]; 4-way partial over j, 64 lanes over d
  const int d = tid & 63;
  const int part = tid >> 6;
  float acc = 0.f;
  for (int j = part; j < NTOK; j += 4)
    acc += s_attn[j] * qkv[(size_t)(b * NTOK + j) * E3 + 2 * E + h * DH + d];
  s_red[tid] = acc;
  __syncthreads();
  if (tid < DH) {
    float o = s_red[tid] + s_red[tid + 64] + s_red[tid + 128] + s_red[tid + 192];
    outh[rowi * E + h * DH + tid] = (_Float16)o;
  }
}

// ------------------------------ head plumbing ------------------------------

__global__ void k_gather_cls(const _Float16* __restrict__ hln, _Float16* __restrict__ cls16) {
  int i = blockIdx.x * 256 + threadIdx.x;
  if (i >= MHEAD * E) return;
  int r = i / E, c = i % E;
  cls16[i] = (r < NB) ? hln[(size_t)(r * NTOK) * E + c] : (_Float16)0.f;
}

__global__ void k_copy_out(const float* __restrict__ src, float* __restrict__ dst) {
  int i = blockIdx.x * 256 + threadIdx.x;
  if (i >= NB * NOUT) return;
  int b = i / NOUT, n = i % NOUT;
  dst[i] = src[(size_t)b * NOUTP + n];
}

// --------------------------------- launch ----------------------------------

extern "C" void kernel_launch(void* const* d_in, const int* in_sizes, int n_in,
                              void* d_out, int out_size, void* d_ws, size_t ws_size,
                              hipStream_t stream) {
  (void)in_sizes; (void)n_in; (void)out_size; (void)ws_size;
  const float* x         = (const float*)d_in[0];
  const float* conv_w    = (const float*)d_in[1];
  const float* conv_b    = (const float*)d_in[2];
  const float* pos_embed = (const float*)d_in[3];
  const float* cls_token = (const float*)d_in[4];
  const float* ln1_w = (const float*)d_in[5];
  const float* ln1_b = (const float*)d_in[6];
  const float* attn_w = (const float*)d_in[7];
  const float* attn_b = (const float*)d_in[8];
  const float* proj_w = (const float*)d_in[9];
  const float* proj_b = (const float*)d_in[10];
  const float* ln2_w = (const float*)d_in[11];
  const float* ln2_b = (const float*)d_in[12];
  const float* fc1_w = (const float*)d_in[13];
  const float* fc1_b = (const float*)d_in[14];
  const float* fc2_w = (const float*)d_in[15];
  const float* fc2_b = (const float*)d_in[16];
  const float* lnf_w = (const float*)d_in[17];
  const float* lnf_b = (const float*)d_in[18];
  const float* head_w = (const float*)d_in[19];
  const float* head_b = (const float*)d_in[20];
  float* out = (float*)d_out;

  // ---- workspace carving (256B aligned) ----
  char* ws = (char*)d_ws;
  size_t off = 0;
  auto alloc = [&](size_t bytes) -> char* {
    char* p = ws + off;
    off = (off + bytes + 255) & ~(size_t)255;
    return p;
  };
  _Float16* hW_qkv  = (_Float16*)alloc((size_t)E3 * E * 2);
  _Float16* hW_proj = (_Float16*)alloc((size_t)E * E * 2);
  _Float16* hW_fc1  = (_Float16*)alloc((size_t)DFF * E * 2);
  _Float16* hW_fc2  = (_Float16*)alloc((size_t)E * DFF * 2);
  _Float16* hW_head = (_Float16*)alloc((size_t)NOUTP * E * 2);
  _Float16* hW_conv = (_Float16*)alloc((size_t)E * E * 2);
  _Float16* hA      = (_Float16*)alloc((size_t)MPAD * E * 2);   // padded rows zero
  _Float16* hLN     = (_Float16*)alloc((size_t)MPAD * E * 2);
  _Float16* hATT    = (_Float16*)alloc((size_t)MPAD * E * 2);
  _Float16* hFC1    = (_Float16*)alloc((size_t)MPAD * DFF * 2);
  _Float16* hCLS    = (_Float16*)alloc((size_t)MHEAD * E * 2);
  float* tokF    = (float*)alloc((size_t)MPAD * E * 4);
  float* h_state = (float*)alloc((size_t)MPAD * E * 4);
  float* qkvF    = (float*)alloc((size_t)MPAD * E3 * 4);
  float* headF   = (float*)alloc((size_t)MHEAD * NOUTP * 4);
  const size_t used_bytes = off;

  // zero all intermediates once per launch (padded rows must be 0)
  k_fill0<<<4096, 256, 0, stream>>>((unsigned int*)d_ws, used_bytes / 4);

  auto gemm = [&](const _Float16* A, const _Float16* Wt, const float* bias, int Nreal,
                  float* oF, _Float16* oH, const float* resid, int K, int Npad,
                  int act, int Mtiles /*even*/) {
    int Ntiles2 = Npad / 32;
    dim3 grid(CEILDIV(Ntiles2, 4), Mtiles / 2);
    k_gemm_wmma<<<grid, 128, 0, stream>>>(A, Wt, bias, Nreal, oF, oH, resid, K, Npad, act);
  };

  // ---- patch embedding ----
  k_im2col<<<CEILDIV(MPATCH * E, 256), 256, 0, stream>>>(x, hA);
  k_cvt<<<CEILDIV(E * E, 256), 256, 0, stream>>>(conv_w, hW_conv, E * E); // already [E][K]
  gemm(hA, hW_conv, conv_b, E, tokF, nullptr, nullptr, E, E, 0, MPAD / 16);
  k_assemble<<<CEILDIV(MROWS * E, 256), 256, 0, stream>>>(tokF, pos_embed, cls_token, h_state);

  // ---- transformer layers ----
  for (int l = 0; l < 2; ++l) {
    const float* aW = attn_w + (size_t)l * E * E3;
    const float* aB = attn_b + (size_t)l * E3;
    const float* pW = proj_w + (size_t)l * E * E;
    const float* pB = proj_b + (size_t)l * E;
    const float* f1W = fc1_w + (size_t)l * E * DFF;
    const float* f1B = fc1_b + (size_t)l * DFF;
    const float* f2W = fc2_w + (size_t)l * DFF * E;
    const float* f2B = fc2_b + (size_t)l * E;

    // attention block
    k_ln<<<MROWS, 256, 0, stream>>>(h_state, ln1_w + (size_t)l * E, ln1_b + (size_t)l * E, hLN);
    k_wt_t<<<CEILDIV(E3 * E, 256), 256, 0, stream>>>(aW, hW_qkv, E, E3, E3 * E);
    gemm(hLN, hW_qkv, aB, E3, qkvF, nullptr, nullptr, E, E3, 0, MPAD / 16);
    k_attn<<<dim3(NTOK, NHD, NB), 256, 0, stream>>>(qkvF, hATT);
    k_wt_t<<<CEILDIV(E * E, 256), 256, 0, stream>>>(pW, hW_proj, E, E, E * E);
    gemm(hATT, hW_proj, pB, E, h_state, nullptr, h_state, E, E, 0, MPAD / 16);

    // MLP block
    k_ln<<<MROWS, 256, 0, stream>>>(h_state, ln2_w + (size_t)l * E, ln2_b + (size_t)l * E, hLN);
    k_wt_t<<<CEILDIV(DFF * E, 256), 256, 0, stream>>>(f1W, hW_fc1, E, DFF, DFF * E);
    gemm(hLN, hW_fc1, f1B, DFF, nullptr, hFC1, nullptr, E, DFF, 1, MPAD / 16);
    k_wt_t<<<CEILDIV(E * DFF, 256), 256, 0, stream>>>(f2W, hW_fc2, DFF, E, E * DFF);
    gemm(hFC1, hW_fc2, f2B, E, h_state, nullptr, h_state, DFF, E, 0, MPAD / 16);
  }

  // ---- final LN + head ----
  k_ln<<<MROWS, 256, 0, stream>>>(h_state, lnf_w, lnf_b, hLN);
  k_gather_cls<<<CEILDIV(MHEAD * E, 256), 256, 0, stream>>>(hLN, hCLS);
  k_wt_t<<<CEILDIV(NOUTP * E, 256), 256, 0, stream>>>(head_w, hW_head, E, NOUT, NOUTP * E);
  gemm(hCLS, hW_head, head_b, NOUT, headF, nullptr, nullptr, E, NOUTP, 0, 2);
  k_copy_out<<<CEILDIV(NB * NOUT, 256), 256, 0, stream>>>(headF, out);
}
